// Model_72722386256233
// MI455X (gfx1250) — matrix operand order repaired and checked by reading, not compiled, not run on hardware
//
#include <hip/hip_runtime.h>
#include <hip/hip_bf16.h>
#include <math.h>

// ---------------------------------------------------------------------------
// Model dims
// ---------------------------------------------------------------------------
constexpr int Bn   = 1024;
constexpr int In   = 36;
constexpr int Cn   = 8;
constexpr int En   = 128;
constexpr int Pn   = 12;
constexpr int L1   = In * En;      // 4608
constexpr int L1p  = L1 / 4;       // 1152
constexpr int L2p  = L1p / 4;      // 288
constexpr int L3p  = L2p / 3;      // 96
constexpr int Hh   = 64;           // LSTM hidden
constexpr int Gn   = 256;          // 4*H gates
constexpr int Fn   = L3p * Hh;     // 6144 flattened LSTM output

typedef _Float16 f16;
typedef _Float16 v16h __attribute__((ext_vector_type(16)));
typedef _Float16 v8h  __attribute__((ext_vector_type(8)));
typedef float    v8f  __attribute__((ext_vector_type(8)));

// ---------------------------------------------------------------------------
// WMMA helpers (CDNA5 v_wmma_f32_16x16x32_f16).
// A-frag (16x32 f16): lane L holds M=L%16, K slots: s<8 -> k0+s+8*(L>=16),
//                     s>=8 -> k0+s+8+8*(L>=16)   (ISA 7.12.2 table)
// B-frag (32x16 f16): lane L holds N=L%16, K slots same as A-frag
// C/D (16x16 f32):    slot r, lane L -> M = r + 8*(L>=16), N = L%16
// ---------------------------------------------------------------------------
__device__ __forceinline__ v8f wmma_f16(v16h a, v16h b, v8f c) {
    return __builtin_amdgcn_wmma_f32_16x16x32_f16(
        /*neg_a=*/false, a, /*neg_b=*/false, b,
        /*c_mod=*/(short)0, c, /*reuse_a=*/false, /*reuse_b=*/false);
}

// p points at tile row 0 of A (row-major, leading dim ld halfs), k0 = K offset.
__device__ __forceinline__ v16h lda_frag(const f16* p, int ld, int k0, int lane) {
    const f16* q = p + (lane & 15) * ld + k0 + ((lane >> 4) << 3);
    v16h a;
#pragma unroll
    for (int s = 0; s < 8; ++s) { a[s] = q[s]; a[s + 8] = q[s + 16]; }
    return a;
}

// p points at B[k0][n0] of a (K x N)-layout matrix with leading dim ldn halfs.
__device__ __forceinline__ v16h ldb_frag(const f16* p, int ldn, int lane) {
    const f16* q = p + (lane & 15) + ((lane >> 4) << 3) * ldn;
    v16h b;
#pragma unroll
    for (int s = 0; s < 8; ++s) { b[s] = q[s * ldn]; b[s + 8] = q[(s + 16) * ldn]; }
    return b;
}

__device__ __forceinline__ float sigm(float v) { return 1.f / (1.f + expf(-v)); }
__device__ __forceinline__ float sshrink(float v) {
    return (v > 0.01f) ? v - 0.01f : ((v < -0.01f) ? v + 0.01f : 0.f);
}

// ---------------------------------------------------------------------------
// 1) Selected ortho-rDFT coefficients S[b,c,m], m=0..35.
//    38-seq [r0,i0,r1,i1,...,r18,i18] -> drop idx1 (i0) -> drop idx19 (r10).
// ---------------------------------------------------------------------------
__global__ __launch_bounds__(256) void k_dft(const float* __restrict__ x,
                                             float* __restrict__ S) {
    int idx = blockIdx.x * 256 + threadIdx.x;
    if (idx >= Bn * Cn * In) return;
    int m  = idx % In;
    int bc = idx / In;
    int c  = bc & (Cn - 1);
    int b  = bc >> 3;
    int j  = (m < 19) ? m : m + 1;          // index into 37-seq
    int k  = (j + 1) >> 1;                  // frequency bin
    bool isReal = (j == 0) || (j & 1);
    float w = 0.17453292519943295f * (float)k; // 2*pi/36 * k
    float s = 0.f;
    for (int i = 0; i < In; ++i) {
        float v   = x[b * (In * Cn) + i * Cn + c];
        float ang = w * (float)i;
        s += v * (isReal ? cosf(ang) : -sinf(ang));
    }
    S[idx] = s * (1.f / 6.f);               // 1/sqrt(36)
}

// ---------------------------------------------------------------------------
// 2) conv1 (Cin=8 -> 8, k=24, SAME pad 11/12) fused with maxpool4.
//    Input synthesized on the fly: in[c][l] = row[c][l>>7] * emb[l&127].
// ---------------------------------------------------------------------------
__global__ __launch_bounds__(256) void k_conv1(const float* __restrict__ base,
                                               int isTime,
                                               const float* __restrict__ emb,
                                               const float* __restrict__ w,
                                               const float* __restrict__ bias,
                                               float* __restrict__ outP) {
    __shared__ float sw[8 * 8 * 24];
    __shared__ float semb[En];
    __shared__ float srow[Cn * In];
    __shared__ float sb[8];
    int b = blockIdx.x, tid = threadIdx.x;
    for (int i = tid; i < 8 * 8 * 24; i += 256) sw[i] = w[i];
    if (tid < En) semb[tid] = emb[tid];
    if (tid < 8)  sb[tid] = bias[tid];
    for (int i = tid; i < Cn * In; i += 256) {
        int c = i / In, ii = i - c * In;
        srow[i] = isTime ? base[b * (In * Cn) + ii * Cn + c]
                         : base[b * (Cn * In) + i];
    }
    __syncthreads();

    for (int p = tid; p < L1p; p += 256) {
        float acc[8][4];
#pragma unroll
        for (int o = 0; o < 8; ++o)
#pragma unroll
            for (int q = 0; q < 4; ++q) acc[o][q] = 0.f;
        int lbase = 4 * p - 11;
        for (int c = 0; c < 8; ++c) {
            float inv[27];
#pragma unroll
            for (int u = 0; u < 27; ++u) {
                int l = lbase + u;
                inv[u] = (l >= 0 && l < L1)
                             ? srow[c * In + (l >> 7)] * semb[l & 127] : 0.f;
            }
            for (int o = 0; o < 8; ++o) {
                const float* wp = sw + o * 192 + c * 24;
#pragma unroll
                for (int q = 0; q < 4; ++q) {
                    float s = acc[o][q];
#pragma unroll
                    for (int t = 0; t < 24; ++t) s += wp[t] * inv[q + t];
                    acc[o][q] = s;
                }
            }
        }
        for (int o = 0; o < 8; ++o) {
            float mx = fmaxf(fmaxf(acc[o][0], acc[o][1]),
                             fmaxf(acc[o][2], acc[o][3]));
            outP[(size_t)b * (8 * L1p) + o * L1p + p] = mx + sb[o];
        }
    }
}

// ---------------------------------------------------------------------------
// 3) conv2 (8 -> 16, k=12, SAME pad 5/6) + maxpool4.
// ---------------------------------------------------------------------------
__global__ __launch_bounds__(256) void k_conv2(const float* __restrict__ P1,
                                               const float* __restrict__ w,
                                               const float* __restrict__ bias,
                                               float* __restrict__ outP) {
    __shared__ float sin1[8 * L1p];       // 36 KB
    __shared__ float sw[16 * 8 * 12];
    __shared__ float sb[16];
    int b = blockIdx.x, tid = threadIdx.x;
    for (int i = tid; i < 8 * L1p; i += 256) sin1[i] = P1[(size_t)b * (8 * L1p) + i];
    for (int i = tid; i < 16 * 8 * 12; i += 256) sw[i] = w[i];
    if (tid < 16) sb[tid] = bias[tid];
    __syncthreads();

    for (int p = tid; p < L2p; p += 256) {
        int lbase = 4 * p - 5;
        for (int oh = 0; oh < 16; oh += 8) {
            float acc[8][4];
#pragma unroll
            for (int o = 0; o < 8; ++o)
#pragma unroll
                for (int q = 0; q < 4; ++q) acc[o][q] = 0.f;
            for (int c = 0; c < 8; ++c) {
                float inv[15];
#pragma unroll
                for (int u = 0; u < 15; ++u) {
                    int l = lbase + u;
                    inv[u] = (l >= 0 && l < L1p) ? sin1[c * L1p + l] : 0.f;
                }
                for (int oo = 0; oo < 8; ++oo) {
                    const float* wp = sw + ((oh + oo) * 8 + c) * 12;
#pragma unroll
                    for (int q = 0; q < 4; ++q) {
                        float s = acc[oo][q];
#pragma unroll
                        for (int t = 0; t < 12; ++t) s += wp[t] * inv[q + t];
                        acc[oo][q] = s;
                    }
                }
            }
            for (int oo = 0; oo < 8; ++oo) {
                float mx = fmaxf(fmaxf(acc[oo][0], acc[oo][1]),
                                 fmaxf(acc[oo][2], acc[oo][3]));
                outP[(size_t)b * (16 * L2p) + (oh + oo) * L2p + p] = mx + sb[oh + oo];
            }
        }
    }
}

// ---------------------------------------------------------------------------
// 4) conv3 (16 -> 32, k=6, SAME pad 2/3) + maxpool3.  Output (B,32,96).
// ---------------------------------------------------------------------------
__global__ __launch_bounds__(128) void k_conv3(const float* __restrict__ P2,
                                               const float* __restrict__ w,
                                               const float* __restrict__ bias,
                                               float* __restrict__ outP) {
    __shared__ float sin2[16 * L2p];      // 18 KB
    __shared__ float sw[32 * 16 * 6];     // 12 KB
    __shared__ float sb[32];
    int b = blockIdx.x, tid = threadIdx.x;
    for (int i = tid; i < 16 * L2p; i += 128) sin2[i] = P2[(size_t)b * (16 * L2p) + i];
    for (int i = tid; i < 32 * 16 * 6; i += 128) sw[i] = w[i];
    if (tid < 32) sb[tid] = bias[tid];
    __syncthreads();

    for (int p = tid; p < L3p; p += 128) {
        int lbase = 3 * p - 2;
        for (int og = 0; og < 32; og += 8) {
            float acc[8][3];
#pragma unroll
            for (int o = 0; o < 8; ++o)
#pragma unroll
                for (int q = 0; q < 3; ++q) acc[o][q] = 0.f;
            for (int c = 0; c < 16; ++c) {
                float inv[8];
#pragma unroll
                for (int u = 0; u < 8; ++u) {
                    int l = lbase + u;
                    inv[u] = (l >= 0 && l < L2p) ? sin2[c * L2p + l] : 0.f;
                }
                for (int oo = 0; oo < 8; ++oo) {
                    const float* wp = sw + ((og + oo) * 16 + c) * 6;
#pragma unroll
                    for (int q = 0; q < 3; ++q) {
                        float s = acc[oo][q];
#pragma unroll
                        for (int t = 0; t < 6; ++t) s += wp[t] * inv[q + t];
                        acc[oo][q] = s;
                    }
                }
            }
            for (int oo = 0; oo < 8; ++oo) {
                float mx = fmaxf(fmaxf(acc[oo][0], acc[oo][1]), acc[oo][2]);
                outP[(size_t)b * (32 * L3p) + (og + oo) * L3p + p] = mx + sb[og + oo];
            }
        }
    }
}

// ---------------------------------------------------------------------------
// 5) Transpose + f32->f16: src (N,K) row-major  ->  dst[k*N + n]  (B layout)
// ---------------------------------------------------------------------------
__global__ __launch_bounds__(256) void k_transpose_cvt(const float* __restrict__ src,
                                                       f16* __restrict__ dst,
                                                       int N, int K) {
    long long idx = (long long)blockIdx.x * 256 + threadIdx.x;
    if (idx >= (long long)N * K) return;
    int n = (int)(idx % N);
    int k = (int)(idx / N);
    dst[idx] = (f16)src[(long long)n * K + k];
}

// ---------------------------------------------------------------------------
// 6) LSTM scan. Each workgroup owns 32 independent batch rows for the whole
//    scan (h in LDS f16, c in registers). Per step, per 16x16 tile:
//    g = bias + x@wihT (1 WMMA, K=32) + h@whhT (2 WMMA, K=64).
//    wihT (32x256) / whhT (64x256) f16 stream from global (L2 resident).
// ---------------------------------------------------------------------------
__global__ __launch_bounds__(256) void k_lstm(
    const float* __restrict__ x, const f16* __restrict__ wihT,
    const f16* __restrict__ whhT, const float* __restrict__ bih,
    const float* __restrict__ bhh, f16* __restrict__ out,
    int nSteps, long long xs_row, long long xs_step, long long xs_k,
    long long os_step, long long os_row) {
    __shared__ f16   s_h[32 * 64];
    __shared__ f16   s_x[32 * 32];
    __shared__ float s_bias[256];
    __shared__ float s_g[32 * 256];       // 32 KB
    const int tid = threadIdx.x, lane = tid & 31, wave = tid >> 5;
    const long long row0 = (long long)blockIdx.x * 32;

    s_bias[tid] = bih[tid] + bhh[tid];
    for (int i = tid; i < 32 * 64; i += 256) s_h[i] = (f16)0.f;
    float creg[8];
#pragma unroll
    for (int i = 0; i < 8; ++i) creg[i] = 0.f;
    __syncthreads();

    for (int t = 0; t < nSteps; ++t) {
        // stage x_t (32 rows x 32 feats) into LDS as f16
        for (int i = tid; i < 32 * 32; i += 256) {
            int r = i >> 5, k = i & 31;
            s_x[i] = (f16)x[(row0 + r) * xs_row + (long long)t * xs_step +
                            (long long)k * xs_k];
        }
        __syncthreads();

        // GEMM: 2 M-tiles x 16 N-tiles, 4 tiles per wave
#pragma unroll
        for (int tt = 0; tt < 4; ++tt) {
            int tile = wave * 4 + tt;
            int mt = tile >> 4, n0 = (tile & 15) * 16;
            v8f acc = {0.f, 0.f, 0.f, 0.f, 0.f, 0.f, 0.f, 0.f};
            v16h a = lda_frag(s_x + mt * 16 * 32, 32, 0, lane);
            v16h bf = ldb_frag(wihT + n0, 256, lane);
            acc = wmma_f16(a, bf, acc);
            a  = lda_frag(s_h + mt * 16 * 64, 64, 0, lane);
            bf = ldb_frag(whhT + n0, 256, lane);
            acc = wmma_f16(a, bf, acc);
            a  = lda_frag(s_h + mt * 16 * 64, 64, 32, lane);
            bf = ldb_frag(whhT + 32 * 256 + n0, 256, lane);
            acc = wmma_f16(a, bf, acc);
            int n  = n0 + (lane & 15);
            int mb = mt * 16 + ((lane >> 4) << 3);
            float bv = s_bias[n];
#pragma unroll
            for (int r = 0; r < 8; ++r) s_g[(mb + r) * 256 + n] = acc[r] + bv;
        }
        __syncthreads();

        // gates: each thread owns 8 fixed (m,hc) cells
#pragma unroll
        for (int i = 0; i < 8; ++i) {
            int e = tid + (i << 8);
            int m = e >> 6, hc = e & 63;
            const float* gp = s_g + m * 256 + hc;
            float cc = sigm(gp[64]) * creg[i] + sigm(gp[0]) * tanhf(gp[128]);
            creg[i] = cc;
            float hv = sigm(gp[192]) * tanhf(cc);
            s_h[m * 64 + hc] = (f16)hv;
            out[(long long)t * os_step + (row0 + m) * os_row + hc] = (f16)hv;
        }
        __syncthreads();
    }
}

// ---------------------------------------------------------------------------
// 7) MLP layer 1: H = leaky(A(1024x6144,f16) @ Bt(6144x256,f16,[k*256+n]) + b)
//    One 16x16 tile per wave, 192 K-steps of v_wmma_f32_16x16x32_f16.
// ---------------------------------------------------------------------------
__global__ __launch_bounds__(256) void k_mlp1(const f16* __restrict__ A,
                                              const f16* __restrict__ Bt,
                                              const float* __restrict__ bias,
                                              float* __restrict__ H) {
    const int tid = threadIdx.x, lane = tid & 31, wave = tid >> 5;
    int tile = blockIdx.x * 8 + wave;
    int mt = tile >> 4, nt = tile & 15;
    const f16* arow = A + (long long)(mt * 16 + (lane & 15)) * Fn + ((lane >> 4) << 3);
    const f16* bcol = Bt + nt * 16 + (lane & 15) + (long long)((lane >> 4) << 3) * 256;
    v8f acc = {0.f, 0.f, 0.f, 0.f, 0.f, 0.f, 0.f, 0.f};
    for (int kt = 0; kt < Fn / 32; ++kt) {
        const f16* ap = arow + kt * 32;
        __builtin_prefetch(ap + 64, 0, 1);
        v8h lo = *(const v8h*)ap;
        v8h hi = *(const v8h*)(ap + 16);
        v16h a;
#pragma unroll
        for (int s = 0; s < 8; ++s) { a[s] = lo[s]; a[s + 8] = hi[s]; }
        const f16* bp = bcol + (long long)kt * 32 * 256;
        v16h b;
#pragma unroll
        for (int s = 0; s < 8; ++s) { b[s] = bp[s * 256]; b[s + 8] = bp[(s + 16) * 256]; }
        acc = wmma_f16(a, b, acc);
    }
    int n  = nt * 16 + (lane & 15);
    int mb = mt * 16 + ((lane >> 4) << 3);
    float bv = bias[n];
#pragma unroll
    for (int r = 0; r < 8; ++r) {
        float v = acc[r] + bv;
        v = (v > 0.f) ? v : 0.01f * v;
        H[(long long)(mb + r) * 256 + n] = v;
    }
}

// ---------------------------------------------------------------------------
// 8) Final: tf/yf (256->12 dots), softshrink + ortho irfft(12), fuse GEMM.
// ---------------------------------------------------------------------------
__global__ __launch_bounds__(64) void k_final(
    const float* __restrict__ Hht, const float* __restrict__ Hhf,
    const float* __restrict__ wt2, const float* __restrict__ bt2,
    const float* __restrict__ wf2, const float* __restrict__ bf2,
    const float* __restrict__ fw, const float* __restrict__ fb,
    float* __restrict__ outp) {
    __shared__ float stf[12], syf[12], sff[12];
    int b = blockIdx.x, tid = threadIdx.x;
    if (tid < 12) {
        float s = bt2[tid];
        const float* hr = Hht + (size_t)b * 256;
        const float* wr = wt2 + tid * 256;
        for (int k = 0; k < 256; ++k) s += hr[k] * wr[k];
        stf[tid] = s;
    } else if (tid < 24) {
        int p = tid - 12;
        float s = bf2[p];
        const float* hr = Hhf + (size_t)b * 256;
        const float* wr = wf2 + p * 256;
        for (int k = 0; k < 256; ++k) s += hr[k] * wr[k];
        syf[p] = s;
    }
    __syncthreads();
    if (tid < 12) {
        int n = tid;
        float re[7], im[7];
#pragma unroll
        for (int k = 0; k < 7; ++k) re[k] = sshrink(syf[k]);
        im[0] = 0.f; im[6] = 0.f;
#pragma unroll
        for (int k = 1; k <= 5; ++k) im[k] = sshrink(syf[6 + k]);
        float s = re[0] + ((n & 1) ? -re[6] : re[6]);
#pragma unroll
        for (int k = 1; k <= 5; ++k) {
            float ang = 0.5235987755982988f * (float)(k * n);   // pi/6*k*n
            s += 2.f * (re[k] * cosf(ang) - im[k] * sinf(ang));
        }
        sff[n] = s * 0.2886751345948129f;                        // 1/sqrt(12)
    }
    __syncthreads();
    if (tid < 12) {
        float o = fb[tid];
#pragma unroll
        for (int q = 0; q < 12; ++q)
            o += fw[tid * 24 + q] * stf[q] + fw[tid * 24 + 12 + q] * sff[q];
        outp[(size_t)b * 12 + tid] = o;
    }
}

// ---------------------------------------------------------------------------
// Host launcher
// ---------------------------------------------------------------------------
extern "C" void kernel_launch(void* const* d_in, const int* in_sizes, int n_in,
                              void* d_out, int out_size, void* d_ws, size_t ws_size,
                              hipStream_t stream) {
    (void)in_sizes; (void)n_in; (void)out_size; (void)ws_size;
    const float* x      = (const float*)d_in[0];
    const float* emb    = (const float*)d_in[2];
    const float* ft_w1  = (const float*)d_in[3];
    const float* ft_b1  = (const float*)d_in[4];
    const float* ft_w2  = (const float*)d_in[5];
    const float* ft_b2  = (const float*)d_in[6];
    const float* ft_w3  = (const float*)d_in[7];
    const float* ft_b3  = (const float*)d_in[8];
    const float* ff_w1  = (const float*)d_in[9];
    const float* ff_b1  = (const float*)d_in[10];
    const float* ff_w2  = (const float*)d_in[11];
    const float* ff_b2  = (const float*)d_in[12];
    const float* ff_w3  = (const float*)d_in[13];
    const float* ff_b3  = (const float*)d_in[14];
    const float* lt_wih = (const float*)d_in[15];
    const float* lt_whh = (const float*)d_in[16];
    const float* lt_bih = (const float*)d_in[17];
    const float* lt_bhh = (const float*)d_in[18];
    const float* lf_wih = (const float*)d_in[19];
    const float* lf_whh = (const float*)d_in[20];
    const float* lf_bih = (const float*)d_in[21];
    const float* lf_bhh = (const float*)d_in[22];
    const float* dt_w1  = (const float*)d_in[23];
    const float* dt_b1  = (const float*)d_in[24];
    const float* dt_w2  = (const float*)d_in[25];
    const float* dt_b2  = (const float*)d_in[26];
    const float* df_w1  = (const float*)d_in[27];
    const float* df_b1  = (const float*)d_in[28];
    const float* df_w2  = (const float*)d_in[29];
    const float* df_b2  = (const float*)d_in[30];
    const float* fuse_w = (const float*)d_in[31];
    const float* fuse_b = (const float*)d_in[32];

    char* ws = (char*)d_ws;
    size_t off = 0;
    auto alloc = [&](size_t bytes) -> void* {
        off = (off + 255) & ~(size_t)255;
        void* p = ws + off;
        off += bytes;
        return p;
    };
    float* S      = (float*)alloc((size_t)Bn * Cn * In * 4);
    float* P1     = (float*)alloc((size_t)Bn * 8 * L1p * 4);
    float* P2     = (float*)alloc((size_t)Bn * 16 * L2p * 4);
    float* P3t    = (float*)alloc((size_t)Bn * 32 * L3p * 4);
    float* P3f    = (float*)alloc((size_t)Bn * 32 * L3p * 4);
    f16*   wihT_t = (f16*)alloc((size_t)32 * 256 * 2);
    f16*   whhT_t = (f16*)alloc((size_t)64 * 256 * 2);
    f16*   wihT_f = (f16*)alloc((size_t)32 * 256 * 2);
    f16*   whhT_f = (f16*)alloc((size_t)64 * 256 * 2);
    f16*   w1T_t  = (f16*)alloc((size_t)Fn * 256 * 2);
    f16*   w1T_f  = (f16*)alloc((size_t)Fn * 256 * 2);
    f16*   yt16   = (f16*)alloc((size_t)Bn * Fn * 2);
    f16*   yf16   = (f16*)alloc((size_t)Bn * Fn * 2);
    float* Hht    = (float*)alloc((size_t)Bn * 256 * 4);
    float* Hhf    = (float*)alloc((size_t)Bn * 256 * 4);

    // frequency-domain coefficients + weight prep
    k_dft<<<(Bn * Cn * In) / 256, 256, 0, stream>>>(x, S);
    k_transpose_cvt<<<(256 * 32 + 255) / 256, 256, 0, stream>>>(lt_wih, wihT_t, 256, 32);
    k_transpose_cvt<<<(256 * 64 + 255) / 256, 256, 0, stream>>>(lt_whh, whhT_t, 256, 64);
    k_transpose_cvt<<<(256 * 32 + 255) / 256, 256, 0, stream>>>(lf_wih, wihT_f, 256, 32);
    k_transpose_cvt<<<(256 * 64 + 255) / 256, 256, 0, stream>>>(lf_whh, whhT_f, 256, 64);
    k_transpose_cvt<<<(256 * Fn + 255) / 256, 256, 0, stream>>>(dt_w1, w1T_t, 256, Fn);
    k_transpose_cvt<<<(256 * Fn + 255) / 256, 256, 0, stream>>>(df_w1, w1T_f, 256, Fn);

    // time branch: convs -> LSTM over B as time (rows j in [0,96), 3 WGs)
    k_conv1<<<Bn, 256, 0, stream>>>(x, 1, emb, ft_w1, ft_b1, P1);
    k_conv2<<<Bn, 256, 0, stream>>>(P1, ft_w2, ft_b2, P2);
    k_conv3<<<Bn, 128, 0, stream>>>(P2, ft_w3, ft_b3, P3t);
    k_lstm<<<L3p / 32, 256, 0, stream>>>(P3t, wihT_t, whhT_t, lt_bih, lt_bhh, yt16,
                                         /*steps*/Bn, /*xs_row*/1, /*xs_step*/3072,
                                         /*xs_k*/96, /*os_step*/Fn, /*os_row*/64);

    // freq branch (reuses P1/P2): convs -> LSTM over 96 steps (rows b, 32 WGs)
    k_conv1<<<Bn, 256, 0, stream>>>(S, 0, emb, ff_w1, ff_b1, P1);
    k_conv2<<<Bn, 256, 0, stream>>>(P1, ff_w2, ff_b2, P2);
    k_conv3<<<Bn, 128, 0, stream>>>(P2, ff_w3, ff_b3, P3f);
    k_lstm<<<Bn / 32, 256, 0, stream>>>(P3f, wihT_f, whhT_f, lf_bih, lf_bhh, yf16,
                                        /*steps*/L3p, /*xs_row*/3072, /*xs_step*/1,
                                        /*xs_k*/96, /*os_step*/64, /*os_row*/Fn);

    // dense heads (WMMA GEMM 1024x256, K=6144) + final fuse
    k_mlp1<<<(Bn / 16) * (256 / 16) / 8, 256, 0, stream>>>(yt16, w1T_t, dt_b1, Hht);
    k_mlp1<<<(Bn / 16) * (256 / 16) / 8, 256, 0, stream>>>(yf16, w1T_f, df_b1, Hhf);
    k_final<<<Bn, 64, 0, stream>>>(Hht, Hhf, dt_w2, dt_b2, df_w2, df_b2,
                                   fuse_w, fuse_b, (float*)d_out);
}
